// MultiHeadAttentionBlock_84189948936393
// MI455X (gfx1250) — compile-verified, matrix operand order checked by reading, not run on hardware
//
#include <hip/hip_runtime.h>
#include <hip/hip_bf16.h>

// ---------------------------------------------------------------------------
// MultiHeadAttention (B=2, S=2048, D=2048, H=16, hd=128) on gfx1250.
// bf16 WMMA pipeline: cvt -> QKV proj GEMM (TDM-staged) -> flash attn -> out.
// ---------------------------------------------------------------------------

#define D_MODEL  2048
#define NUM_HEADS 16
#define HEAD_DIM 128
#define BATCH    2
#define SEQ      2048
#define MROWS    (BATCH * SEQ)   // 4096 total token rows

#define TILE_K    32             // k-slice per TDM transfer / WMMA step
#define TILE_ROWS 128            // block tile rows (M and N)
#define LDS_LD    40             // padded LDS row stride (elems): 80B rows ->
                                 // 16B-aligned chunks, conflict-free b128 reads

typedef __bf16 bf16t;
typedef __attribute__((ext_vector_type(8)))  __bf16 v8bf;
typedef __attribute__((ext_vector_type(16))) __bf16 v16bf;
typedef __attribute__((ext_vector_type(8)))  float  v8f;
typedef __attribute__((ext_vector_type(4)))  unsigned int v4u;
typedef __attribute__((ext_vector_type(4)))  int v4i;
typedef __attribute__((ext_vector_type(8)))  int v8i;

// ---------------------------------------------------------------------------
// Fragment loader for 16x32 bf16 A-operand (and, by symmetry, 32x16 B-operand)
// from a row-major matrix with row stride `ld` (elements).
// Per-lane layout (CDNA5 ISA 7.12.2, 16-bit A 16x32):
//   lanes 0-15 : row = lane,    K = [0..7]  and [16..23]
//   lanes 16-31: row = lane-16, K = [8..15] and [24..31]
// ---------------------------------------------------------------------------
static __device__ __forceinline__ v16bf load_frag(const bf16t* base, int ld) {
  const int lane = threadIdx.x & 31;
  const int r    = lane & 15;
  const int hf   = (lane >> 4) & 1;
  const bf16t* p = base + (size_t)r * ld + hf * 8;
  v8bf lo = *reinterpret_cast<const v8bf*>(p);
  v8bf hi = *reinterpret_cast<const v8bf*>(p + 16);
  return __builtin_shufflevector(lo, hi, 0,1,2,3,4,5,6,7,8,9,10,11,12,13,14,15);
}

static __device__ __forceinline__ v8f wmma_bf16(v16bf a, v16bf b, v8f c) {
  return __builtin_amdgcn_wmma_f32_16x16x32_bf16(false, a, false, b,
                                                 (short)0, c, false, false);
}

// ---------------------------------------------------------------------------
// Tensor Data Mover: DMA one 2D bf16 tile (TILE_ROWS x TILE_K) from global
// into LDS, padding each 64B row out to 80B (pad_interval=16 DW, amount=4 DW)
// so LDS rows land at stride LDS_LD elements.
// D# layout per CDNA5 ISA ch.8 (group0 128b, group1 256b; groups 2/3 unused
// for 2D tiles and passed as zeros).
// ---------------------------------------------------------------------------
static __device__ __forceinline__ void tdm_load_tile(const bf16t* gsrc,
                                                     const bf16t* lds_dst,
                                                     int tensor_rows,
                                                     int row_stride_elems) {
  const unsigned long long ga = (unsigned long long)gsrc;
  const unsigned int lds_addr = (unsigned int)(unsigned long long)lds_dst;

  v4u g0;
  g0[0] = 1u;                                   // count=1, user descriptor
  g0[1] = lds_addr;                             // LDS byte address
  g0[2] = (unsigned int)ga;                     // global_addr[31:0]
  g0[3] = (unsigned int)((ga >> 32) & 0x1FFFFFFu) | (2u << 30); // [56:32]|type=2

  const unsigned int d0 = (unsigned int)row_stride_elems;  // tensor_dim0
  const unsigned int d1 = (unsigned int)tensor_rows;       // tensor_dim1
  v8i g1;
  g1[0] = (int)((1u << 16)        // data_size = 1 -> 2 bytes/elem
              | (1u << 20)        // pad_enable
              | (3u << 22)        // pad_interval: 16 DWORDs (=64B = one row)
              | (3u << 25));      // pad_amount: 4 DWORDs (=16B = 8 elems)
  g1[1] = (int)((d0 & 0xFFFFu) << 16);                       // dim0[15:0]
  g1[2] = (int)(((d0 >> 16) & 0xFFFFu) | ((d1 & 0xFFFFu) << 16));
  g1[3] = (int)(((d1 >> 16) & 0xFFFFu) | ((unsigned)TILE_K << 16)); // tile_dim0
  g1[4] = (int)(TILE_ROWS & 0xFFFF);                         // tile_dim1
  g1[5] = (int)row_stride_elems;                             // dim0_stride lo32
  g1[6] = 0;
  g1[7] = 0;

  v4i z4 = {0, 0, 0, 0};
#if defined(__clang_major__) && (__clang_major__ >= 23)
  v8i z8 = {0, 0, 0, 0, 0, 0, 0, 0};
  __builtin_amdgcn_tensor_load_to_lds(g0, g1, z4, z4, z8, 0);
#else
  __builtin_amdgcn_tensor_load_to_lds(g0, g1, z4, z4, 0);
#endif
}

// ---------------------------------------------------------------------------
// fp32 -> bf16 elementwise conversion
// ---------------------------------------------------------------------------
__global__ void mha_cvt_bf16_kernel(const float* __restrict__ src,
                                    bf16t* __restrict__ dst, int n) {
  int i = blockIdx.x * blockDim.x + threadIdx.x;
  if (i < n) dst[i] = (bf16t)src[i];
}

// ---------------------------------------------------------------------------
// Tiled bf16 GEMM:  y[m,n] = sum_k A[m,k] * W[n,k] + bias[n]
// Block tile 128x128, 8 waves (each 32x64 = 2x4 WMMA fragments).
// A and B 128x32 k-slices are DMA'd into double-buffered LDS by the TDM
// (wave 0 issues, TENSORcnt + barrier gates the swap), all waves consume
// via ds_load_b128.  Epilogue scatter controlled by `mode`:
//   mode 0: Q  -> bf16 [b,h,s,d]
//   mode 1: K  -> bf16 [b,h,s,d]
//   mode 2: V  -> bf16 [b,h,d,s]   (transposed for the PV matmul)
//   mode 3: out-> fp32 [m,n]       (final projection into d_out)
// ---------------------------------------------------------------------------
__global__ __launch_bounds__(256)
void mha_gemm_bf16_kernel(const bf16t* __restrict__ A,
                          const bf16t* __restrict__ W,
                          const float* __restrict__ bias,
                          void* __restrict__ dstv,
                          int M, int N, int K, int mode) {
  __shared__ __align__(16) bf16t atile[2][TILE_ROWS * LDS_LD];
  __shared__ __align__(16) bf16t btile[2][TILE_ROWS * LDS_LD];

  const int w    = threadIdx.x >> 5;
  const int lane = threadIdx.x & 31;
  const int mb = blockIdx.x * 128;          // block row base
  const int nb = blockIdx.y * 128;          // block col base
  const int wm = (w >> 1) * 32;             // wave row offset in block
  const int wn = (w & 1) * 64;              // wave col offset in block

  v8f acc[2][4];
  #pragma unroll
  for (int i = 0; i < 2; ++i)
    #pragma unroll
    for (int j = 0; j < 4; ++j) acc[i][j] = {};

  // Prologue: DMA first A/B k-slices into buffer 0.
  if (w == 0) {
    tdm_load_tile(A + (size_t)mb * K, atile[0], M, K);
    tdm_load_tile(W + (size_t)nb * K, btile[0], N, K);
  }

  int buf = 0;
  for (int k = 0; k < K; k += TILE_K) {
    if (w == 0) {
      if (k + TILE_K < K) {
        // Prefetch next slices into the other buffer, then wait for the
        // current pair (TDM completes in order per wave: <=2 outstanding).
        tdm_load_tile(A + (size_t)mb * K + (k + TILE_K), atile[buf ^ 1], M, K);
        tdm_load_tile(W + (size_t)nb * K + (k + TILE_K), btile[buf ^ 1], N, K);
        __builtin_amdgcn_s_wait_tensorcnt(2);
      } else {
        __builtin_amdgcn_s_wait_tensorcnt(0);
      }
    }
    __syncthreads();                         // tiles in `buf` ready for all

    v16bf a0 = load_frag(&atile[buf][(wm +  0) * LDS_LD], LDS_LD);
    v16bf a1 = load_frag(&atile[buf][(wm + 16) * LDS_LD], LDS_LD);
    v16bf b0 = load_frag(&btile[buf][(wn +  0) * LDS_LD], LDS_LD);
    v16bf b1 = load_frag(&btile[buf][(wn + 16) * LDS_LD], LDS_LD);
    v16bf b2 = load_frag(&btile[buf][(wn + 32) * LDS_LD], LDS_LD);
    v16bf b3 = load_frag(&btile[buf][(wn + 48) * LDS_LD], LDS_LD);
    acc[0][0] = wmma_bf16(a0, b0, acc[0][0]);
    acc[0][1] = wmma_bf16(a0, b1, acc[0][1]);
    acc[0][2] = wmma_bf16(a0, b2, acc[0][2]);
    acc[0][3] = wmma_bf16(a0, b3, acc[0][3]);
    acc[1][0] = wmma_bf16(a1, b0, acc[1][0]);
    acc[1][1] = wmma_bf16(a1, b1, acc[1][1]);
    acc[1][2] = wmma_bf16(a1, b2, acc[1][2]);
    acc[1][3] = wmma_bf16(a1, b3, acc[1][3]);

    __syncthreads();                         // done reading before overwrite
    buf ^= 1;
  }

  // Epilogue. C/D layout: col n = lane&15 ; row m = elem + 8*(lane>>4).
  const int c  = lane & 15;
  const int hf = lane >> 4;
  #pragma unroll
  for (int j = 0; j < 4; ++j) {
    const int n  = nb + wn + j * 16 + c;
    const float bv = bias[n];
    #pragma unroll
    for (int i = 0; i < 2; ++i) {
      #pragma unroll
      for (int e = 0; e < 8; ++e) {
        const int m = mb + wm + i * 16 + e + 8 * hf;
        const float v = acc[i][j][e] + bv;
        if (mode == 3) {
          ((float*)dstv)[(size_t)m * N + n] = v;
        } else {
          const int bb = m >> 11;          // m / SEQ
          const int s  = m & (SEQ - 1);
          const int hh = n >> 7;           // n / HEAD_DIM
          const int d  = n & (HEAD_DIM - 1);
          size_t idx;
          if (mode == 2)
            idx = ((size_t)(bb * NUM_HEADS + hh) * HEAD_DIM + d) * SEQ + s;
          else
            idx = ((size_t)(bb * NUM_HEADS + hh) * SEQ + s) * HEAD_DIM + d;
          ((bf16t*)dstv)[idx] = (bf16t)v;
        }
      }
    }
  }
}

// ---------------------------------------------------------------------------
// Flash attention: one wave (32 threads) handles 16 query rows of one (b,h).
// Q,K bf16 [b,h,s,d] ; Vt bf16 [b,h,d,s] ; output ctx bf16 [b,s,h*128+d].
// Online softmax; 32 keys per iteration (two 16x16 score tiles).
// P converted C-layout -> A-layout through a 16x32 bf16 LDS tile.
// ---------------------------------------------------------------------------
__global__ __launch_bounds__(32)
void mha_flash_attn_kernel(const bf16t* __restrict__ Q,
                           const bf16t* __restrict__ Kmat,
                           const bf16t* __restrict__ Vt,
                           bf16t* __restrict__ Cout) {
  __shared__ __align__(16) bf16t pbuf[16 * 32];

  const int lane = threadIdx.x & 31;
  const int bh = blockIdx.y;              // 0..31
  const int q0 = blockIdx.x * 16;         // query tile base row
  const int b  = bh >> 4;
  const int h  = bh & 15;

  const bf16t* Qh = Q    + (size_t)bh * SEQ * HEAD_DIM;
  const bf16t* Kh = Kmat + (size_t)bh * SEQ * HEAD_DIM;
  const bf16t* Vh = Vt   + (size_t)bh * HEAD_DIM * SEQ;

  v16bf qf[4];
  #pragma unroll
  for (int kk = 0; kk < 4; ++kk)
    qf[kk] = load_frag(Qh + (size_t)q0 * HEAD_DIM + kk * 32, HEAD_DIM);

  v8f ctx[8];
  #pragma unroll
  for (int f = 0; f < 8; ++f) ctx[f] = {};

  float mstat[8], lstat[8];
  #pragma unroll
  for (int e = 0; e < 8; ++e) { mstat[e] = -1e30f; lstat[e] = 0.0f; }

  const int c  = lane & 15;
  const int hf = lane >> 4;
  const float sc = 0.088388347648318447f;  // 1/sqrt(HEAD_DIM)

  for (int kb = 0; kb < q0 + 16; kb += 32) {
    // ---- S = Q * K^T for 32 keys (two 16-wide score fragments) ----
    v8f s0 = {}, s1 = {};
    #pragma unroll
    for (int kk = 0; kk < 4; ++kk) {
      v16bf k0 = load_frag(Kh + (size_t)(kb)      * HEAD_DIM + kk * 32, HEAD_DIM);
      v16bf k1 = load_frag(Kh + (size_t)(kb + 16) * HEAD_DIM + kk * 32, HEAD_DIM);
      s0 = wmma_bf16(qf[kk], k0, s0);
      s1 = wmma_bf16(qf[kk], k1, s1);
    }

    // ---- scale, causal mask, online softmax (per C-fragment row) ----
    #pragma unroll
    for (int e = 0; e < 8; ++e) {
      const int m = q0 + e + 8 * hf;           // global query row
      float v0 = s0[e] * sc;
      float v1 = s1[e] * sc;
      if (kb + c      > m) v0 = -1e30f;        // causal: key > query
      if (kb + 16 + c > m) v1 = -1e30f;

      float rm = fmaxf(v0, v1);
      #pragma unroll
      for (int off = 8; off > 0; off >>= 1)
        rm = fmaxf(rm, __shfl_xor(rm, off, 16));

      const float mnew   = fmaxf(mstat[e], rm);
      const float pscale = __expf(mstat[e] - mnew);
      mstat[e] = mnew;

      v0 = __expf(v0 - mnew);
      v1 = __expf(v1 - mnew);

      float rs = v0 + v1;
      #pragma unroll
      for (int off = 8; off > 0; off >>= 1)
        rs += __shfl_xor(rs, off, 16);
      lstat[e] = lstat[e] * pscale + rs;

      #pragma unroll
      for (int f = 0; f < 8; ++f) ctx[f][e] *= pscale;

      pbuf[(e + 8 * hf) * 32 + c]      = (bf16t)v0;
      pbuf[(e + 8 * hf) * 32 + 16 + c] = (bf16t)v1;
    }
    __syncthreads();
    v16bf pf = load_frag(pbuf, 32);            // re-read as A-layout fragment
    __syncthreads();

    // ---- ctx += P * V  (B fragment = contiguous rows of transposed V) ----
    #pragma unroll
    for (int f = 0; f < 8; ++f) {
      v16bf vf = load_frag(Vh + (size_t)(f * 16) * SEQ + kb, SEQ);
      ctx[f] = wmma_bf16(pf, vf, ctx[f]);
    }
  }

  // ---- finalize: divide by row sum, store ctx as [b, s, h*128 + d] bf16 ----
  #pragma unroll
  for (int e = 0; e < 8; ++e) {
    const float inv = 1.0f / lstat[e];
    const int srow = q0 + e + 8 * hf;
    #pragma unroll
    for (int f = 0; f < 8; ++f) {
      const float v = ctx[f][e] * inv;
      Cout[(size_t)(b * SEQ + srow) * D_MODEL + h * HEAD_DIM + f * 16 + c] =
          (bf16t)v;
    }
  }
}

// ---------------------------------------------------------------------------
// Host launcher
// ---------------------------------------------------------------------------
extern "C" void kernel_launch(void* const* d_in, const int* in_sizes, int n_in,
                              void* d_out, int out_size, void* d_ws, size_t ws_size,
                              hipStream_t stream) {
  (void)in_sizes; (void)n_in; (void)out_size; (void)ws_size;

  const float* x  = (const float*)d_in[0];
  const float* wq = (const float*)d_in[1];
  const float* bq = (const float*)d_in[2];
  const float* wk = (const float*)d_in[3];
  const float* bk = (const float*)d_in[4];
  const float* wv = (const float*)d_in[5];
  const float* bv = (const float*)d_in[6];
  const float* wo = (const float*)d_in[7];
  const float* bo = (const float*)d_in[8];

  const size_t NX = (size_t)MROWS * D_MODEL;     // 8,388,608
  const size_t NW = (size_t)D_MODEL * D_MODEL;   // 4,194,304

  bf16t* ws  = (bf16t*)d_ws;                     // ~117 MB of bf16 scratch
  bf16t* xb  = ws;
  bf16t* wqb = xb  + NX;
  bf16t* wkb = wqb + NW;
  bf16t* wvb = wkb + NW;
  bf16t* wob = wvb + NW;
  bf16t* Qb  = wob + NW;   // [b,h,s,d]
  bf16t* Kb  = Qb  + NX;   // [b,h,s,d]
  bf16t* Vtb = Kb  + NX;   // [b,h,d,s]
  bf16t* Cb  = Vtb + NX;   // [b,s,h*128+d]

  // 1) fp32 -> bf16 conversions
  {
    const int T = 256;
    mha_cvt_bf16_kernel<<<(int)((NX + T - 1) / T), T, 0, stream>>>(x,  xb,  (int)NX);
    mha_cvt_bf16_kernel<<<(int)((NW + T - 1) / T), T, 0, stream>>>(wq, wqb, (int)NW);
    mha_cvt_bf16_kernel<<<(int)((NW + T - 1) / T), T, 0, stream>>>(wk, wkb, (int)NW);
    mha_cvt_bf16_kernel<<<(int)((NW + T - 1) / T), T, 0, stream>>>(wv, wvb, (int)NW);
    mha_cvt_bf16_kernel<<<(int)((NW + T - 1) / T), T, 0, stream>>>(wo, wob, (int)NW);
  }

  // 2) Q/K/V projections (TDM-staged WMMA GEMM, scattered epilogues)
  dim3 gg(MROWS / 128, D_MODEL / 128);  // 32 x 16
  mha_gemm_bf16_kernel<<<gg, 256, 0, stream>>>(xb, wqb, bq, Qb,
                                               MROWS, D_MODEL, D_MODEL, 0);
  mha_gemm_bf16_kernel<<<gg, 256, 0, stream>>>(xb, wkb, bk, Kb,
                                               MROWS, D_MODEL, D_MODEL, 1);
  mha_gemm_bf16_kernel<<<gg, 256, 0, stream>>>(xb, wvb, bv, Vtb,
                                               MROWS, D_MODEL, D_MODEL, 2);

  // 3) causal flash attention: one wave per 16 query rows per (b,h)
  mha_flash_attn_kernel<<<dim3(SEQ / 16, BATCH * NUM_HEADS), 32, 0, stream>>>(
      Qb, Kb, Vtb, Cb);

  // 4) output projection -> fp32 d_out
  mha_gemm_bf16_kernel<<<gg, 256, 0, stream>>>(Cb, wob, bo, (float*)d_out,
                                               MROWS, D_MODEL, D_MODEL, 3);
}
